// MultiHeadAttention_21938692948516
// MI455X (gfx1250) — matrix operand order repaired and checked by reading, not compiled, not run on hardware
//
#include <hip/hip_runtime.h>

// MHA: B=2, S=2048, D=1024, H=16, DK=64.
// All matmuls on the CDNA5 matrix pipe: v_wmma_f32_16x16x32_bf16.
// K/V tiles double-buffered via GLOBAL_LOAD_ASYNC_TO_LDS_B128; softmax row
// sums computed with a ones-matrix WMMA instead of ds_bpermute reductions.
#define BATCH 2
#define SEQ   2048
#define DMODEL 1024
#define NHEAD 16
#define DK    64

typedef __attribute__((ext_vector_type(16))) __bf16 v16bf;
typedef __attribute__((ext_vector_type(8)))  float  v8f;
typedef __attribute__((ext_vector_type(4)))  int    v4i;

union BF16x16 {
  v16bf v;
  unsigned int u[8];
};
union Pack8 {
  uint4 q;
  unsigned short h[8];
};

#if __has_builtin(__builtin_amdgcn_global_load_async_to_lds_b128) && \
    __has_builtin(__builtin_amdgcn_s_wait_asynccnt)
#define USE_ASYNC 1
#else
#define USE_ASYNC 0
#endif

#if USE_ASYNC
typedef __attribute__((address_space(1))) v4i gv4i;   // global int4
typedef __attribute__((address_space(3))) v4i lv4i;   // LDS int4
// 16-byte global -> LDS async copy (tracked by ASYNCcnt)
__device__ __forceinline__ void async_copy16(const void* g, void* l) {
  __builtin_amdgcn_global_load_async_to_lds_b128((gv4i*)g, (lv4i*)l, 0, 0);
}
#endif

__device__ __forceinline__ unsigned short f32_to_bf16(float f) {
  unsigned int u = __float_as_uint(f);
  u = (u + 0x7FFFu + ((u >> 16) & 1u)) >> 16;   // round-to-nearest-even
  return (unsigned short)u;
}

__device__ __forceinline__ v8f wmma_bf16(v16bf a, v16bf b, v8f c) {
  return __builtin_amdgcn_wmma_f32_16x16x32_bf16(false, a, false, b, (short)0, c,
                                                 false, false);
}

__device__ __forceinline__ v16bf ones_bf16() {
  BF16x16 f;
#pragma unroll
  for (int i = 0; i < 8; ++i) f.u[i] = 0x3F803F80u;   // {1.0bf16, 1.0bf16}
  return f.v;
}

// A fragment (16x32 bf16, MxK): lane holds row m = row_base + (lane&15);
// element i -> k = k0 + (i<8?0:16) + 8*(lane>=16) + (i&7). Pairs are k-contiguous.
template <int LD>
__device__ __forceinline__ v16bf load_frag_a(const unsigned short* __restrict__ s,
                                             int row_base, int k0) {
  const int lane = threadIdx.x & 31;
  const int m  = row_base + (lane & 15);
  const int kb = (lane >> 4) << 3;                 // 0 or 8
  const unsigned short* p = s + m * LD + k0 + kb;
  BF16x16 f;
#pragma unroll
  for (int i = 0; i < 4; ++i) f.u[i]     = *(const unsigned int*)(p + 2 * i);
#pragma unroll
  for (int i = 0; i < 4; ++i) f.u[4 + i] = *(const unsigned int*)(p + 16 + 2 * i);
  return f.v;
}

// B fragment (32x16 bf16, KxN): lane holds col n = n_base + (lane&15);
// element i -> k = k0 + (i<8?0:16) + 8*(lane>=16) + (i&7). Stored k-contiguous per n-row.
template <int LD>
__device__ __forceinline__ v16bf load_frag_b(const unsigned short* __restrict__ s,
                                             int n_base, int k0) {
  const int lane = threadIdx.x & 31;
  const int n  = n_base + (lane & 15);
  const int kb = (lane >> 4) << 3;                 // 0 or 8
  const unsigned short* p = s + n * LD + k0 + kb;
  BF16x16 f;
#pragma unroll
  for (int i = 0; i < 4; ++i) f.u[i]     = *(const unsigned int*)(p + 2 * i);
#pragma unroll
  for (int i = 0; i < 4; ++i) f.u[4 + i] = *(const unsigned int*)(p + 16 + 2 * i);
  return f.v;
}

// ---------------------------------------------------------------------------
// Kernel 1: projection. TRANS=0: out[b,h,s,dk] bf16. TRANS=1: out[b,h,dk,s]
// (used for V so the attention V-tile is a straight 16B-chunk copy).
// grid = (M/128, D/64); block = 256 (8 waves); wave computes 16 x 64.
// ---------------------------------------------------------------------------
template <bool TRANS>
__global__ __launch_bounds__(256) void proj_bf16_kernel(
    const float* __restrict__ X,      // [B*S, D] fp32
    const float* __restrict__ W,      // [D, D] fp32 (out, in)
    const float* __restrict__ bias,   // [D]
    unsigned short* __restrict__ out)
{
  __shared__ __align__(16) unsigned short As[128][72];
  __shared__ __align__(16) unsigned short Bs[64][72];

  const int tid  = threadIdx.x;
  const int wave = tid >> 5;
  const int lane = tid & 31;
  const int m0   = blockIdx.x * 128;
  const int h    = blockIdx.y;          // 64-column tile == head

  v8f acc[4] = {};

  for (int k0 = 0; k0 < DMODEL; k0 += 64) {
    __syncthreads();
    for (int idx = tid; idx < 128 * 64; idx += 256) {
      int r = idx >> 6, c = idx & 63;
      As[r][c] = f32_to_bf16(X[(size_t)(m0 + r) * DMODEL + k0 + c]);
    }
    for (int idx = tid; idx < 64 * 64; idx += 256) {
      int r = idx >> 6, c = idx & 63;
      Bs[r][c] = f32_to_bf16(W[(size_t)(h * 64 + r) * DMODEL + k0 + c]);
    }
    __syncthreads();

#pragma unroll
    for (int kk = 0; kk < 64; kk += 32) {
      v16bf a = load_frag_a<72>(&As[0][0], wave * 16, kk);
#pragma unroll
      for (int nt = 0; nt < 4; ++nt)
        acc[nt] = wmma_bf16(a, load_frag_b<72>(&Bs[0][0], nt * 16, kk), acc[nt]);
    }
  }

  const int col   = lane & 15;
  const int rbase = (lane >> 4) * 8;
  const int b_    = m0 >> 11;                       // tile never straddles batch
  const int srow  = (m0 & (SEQ - 1)) + wave * 16 + rbase;

  if (!TRANS) {
#pragma unroll
    for (int nt = 0; nt < 4; ++nt) {
      const int n  = h * 64 + nt * 16 + col;
      const float bv = bias[n];
#pragma unroll
      for (int r = 0; r < 8; ++r) {
        out[(((size_t)b_ * NHEAD + h) * SEQ + (srow + r)) * DK + (nt * 16 + col)] =
            f32_to_bf16(acc[nt][r] + bv);
      }
    }
  } else {
    // [B,H,DK,S]: lane's 8 rows are s-consecutive -> one 16B store per n-tile
#pragma unroll
    for (int nt = 0; nt < 4; ++nt) {
      const int dk = nt * 16 + col;
      const float bv = bias[h * 64 + dk];
      Pack8 pk;
#pragma unroll
      for (int r = 0; r < 8; ++r) pk.h[r] = f32_to_bf16(acc[nt][r] + bv);
      *(uint4*)&out[(((size_t)b_ * NHEAD + h) * DK + dk) * SEQ + srow] = pk.q;
    }
  }
}

// ---------------------------------------------------------------------------
// Kernel 2: flash attention per (b,h). grid = (S/128, B*H); block = 256.
// Wave owns a 16-row Q tile; block streams 64-key K/V tiles, double-buffered
// via async global->LDS copies (4 x 16B chunks per thread per tile).
// Per tile: 8 QK^T WMMAs + 2 row-sum WMMAs + 8 PV WMMAs; only the row-max
// needs cross-lane shuffles.
// ---------------------------------------------------------------------------
__global__ __launch_bounds__(256) void flash_attn_kernel(
    const unsigned short* __restrict__ Qp,   // [B,H,S,DK] bf16
    const unsigned short* __restrict__ Kp,   // [B,H,S,DK] bf16
    const unsigned short* __restrict__ Vt,   // [B,H,DK,S] bf16 (transposed)
    const int* __restrict__ mask,            // [S,S]
    unsigned short* __restrict__ Ctx)        // [B,S,D] bf16
{
  __shared__ __align__(16) unsigned short Ks[2][64][72];   // [key][dk]
  __shared__ __align__(16) unsigned short Vs[2][64][72];   // [dk][key]
  __shared__ __align__(16) unsigned short Ps[8][16][72];   // per-wave P / ctx stage

  const int tid  = threadIdx.x;
  const int wave = tid >> 5;
  const int lane = tid & 31;
  const int bh   = blockIdx.y;
  const int q0   = blockIdx.x * 128 + wave * 16;

  const unsigned short* Q = Qp + (size_t)bh * SEQ * DK;
  const unsigned short* K = Kp + (size_t)bh * SEQ * DK;
  const unsigned short* V = Vt + (size_t)bh * SEQ * DK;   // [DK][SEQ]

  const v16bf qa0 = load_frag_a<DK>(Q, q0, 0);
  const v16bf qa1 = load_frag_a<DK>(Q, q0, 32);
  const v16bf one = ones_bf16();

  auto issue_tile = [&](int kt, int buf) {
#if USE_ASYNC
#pragma unroll
    for (int j = 0; j < 2; ++j) {   // K tile: 64 rows x 128B -> 512 chunks
      const int item = j * 256 + tid;
      const int r = item >> 3, c = item & 7;
      async_copy16(&K[(size_t)(kt + r) * DK + c * 8], &Ks[buf][r][c * 8]);
    }
#pragma unroll
    for (int j = 0; j < 2; ++j) {   // V tile: 64 dk-rows x 128B -> 512 chunks
      const int item = j * 256 + tid;
      const int d = item >> 3, c = item & 7;
      async_copy16(&V[(size_t)d * SEQ + kt + c * 8], &Vs[buf][d][c * 8]);
    }
#else
    for (int idx = tid; idx < 64 * 64; idx += 256) {
      int r = idx >> 6, c = idx & 63;
      Ks[buf][r][c] = K[(size_t)(kt + r) * DK + c];
      Vs[buf][r][c] = V[(size_t)r * SEQ + kt + c];
    }
#endif
  };

  v8f ctx[4] = {};
  float m_i[8], l_i[8];
#pragma unroll
  for (int r = 0; r < 8; ++r) { m_i[r] = -3.0e38f; l_i[r] = 0.0f; }

  const int col   = lane & 15;
  const int rbase = (lane >> 4) * 8;
  const float scale = 0.125f;                  // 1/sqrt(64)

  issue_tile(0, 0);
  int cur = 0;
  for (int kt = 0; kt < SEQ; kt += 64) {
    const bool havenext = (kt + 64) < SEQ;
    if (havenext) issue_tile(kt + 64, cur ^ 1);   // prefetch next tile
#if USE_ASYNC
    if (havenext) __builtin_amdgcn_s_wait_asynccnt(4);  // this tile's 4 chunks done
    else          __builtin_amdgcn_s_wait_asynccnt(0);
#endif
    __syncthreads();

    // scores: four 16x16 tiles covering 64 keys
    float sv[4][8];
#pragma unroll
    for (int ct = 0; ct < 4; ++ct) {
      v8f sc = {};
      sc = wmma_bf16(qa0, load_frag_b<72>(&Ks[cur][0][0], ct * 16, 0),  sc);
      sc = wmma_bf16(qa1, load_frag_b<72>(&Ks[cur][0][0], ct * 16, 32), sc);
#pragma unroll
      for (int r = 0; r < 8; ++r) {
        const int qq = q0 + rbase + r;
        const int kk = kt + ct * 16 + col;
        const float x = sc[r] * scale;
        sv[ct][r] = (mask[(size_t)qq * SEQ + kk] == 0) ? -1.0e9f : x;
      }
    }

    // online softmax: row max via shuffles; exp; P to wave-private LDS
    float alpha[8];
#pragma unroll
    for (int r = 0; r < 8; ++r) {
      float lm = fmaxf(fmaxf(sv[0][r], sv[1][r]), fmaxf(sv[2][r], sv[3][r]));
#pragma unroll
      for (int off = 8; off >= 1; off >>= 1)
        lm = fmaxf(lm, __shfl_xor(lm, off, 32));
      const float mnew = fmaxf(m_i[r], lm);
      alpha[r] = __expf(m_i[r] - mnew);
      m_i[r] = mnew;
#pragma unroll
      for (int ct = 0; ct < 4; ++ct)
        Ps[wave][rbase + r][ct * 16 + col] = f32_to_bf16(__expf(sv[ct][r] - mnew));
#pragma unroll
      for (int nt = 0; nt < 4; ++nt) ctx[nt][r] *= alpha[r];
    }

    // P fragments (16x64 across two 16x32 A-frags), wave-private -> in-order DS
    const v16bf pf0 = load_frag_a<72>(&Ps[wave][0][0], 0, 0);
    const v16bf pf1 = load_frag_a<72>(&Ps[wave][0][0], 0, 32);

    // row sums on the matrix pipe: rs = P x ones (replicated across lanes)
    v8f rs = {};
    rs = wmma_bf16(pf0, one, rs);
    rs = wmma_bf16(pf1, one, rs);
#pragma unroll
    for (int r = 0; r < 8; ++r) l_i[r] = l_i[r] * alpha[r] + rs[r];

    // ctx += P (16x64) x V (64x64)
#pragma unroll
    for (int nt = 0; nt < 4; ++nt) {
      ctx[nt] = wmma_bf16(pf0, load_frag_b<72>(&Vs[cur][0][0], nt * 16, 0),  ctx[nt]);
      ctx[nt] = wmma_bf16(pf1, load_frag_b<72>(&Vs[cur][0][0], nt * 16, 32), ctx[nt]);
    }

    __syncthreads();
    cur ^= 1;
  }

  // normalize, stage through wave-private LDS, 16B-coalesced stores
#pragma unroll
  for (int r = 0; r < 8; ++r) {
    const float inv = 1.0f / l_i[r];
#pragma unroll
    for (int nt = 0; nt < 4; ++nt)
      Ps[wave][rbase + r][nt * 16 + col] = f32_to_bf16(ctx[nt][r] * inv);
  }
  const int b_ = bh >> 4;
  const int h_ = bh & 15;
#pragma unroll
  for (int j = 0; j < 4; ++j) {
    const int item = j * 32 + lane;          // 128 chunks of 16B per wave
    const int r = item & 15;
    const int c = item >> 4;                 // 0..7
    uint4 val = *(const uint4*)&Ps[wave][r][c * 8];
    *(uint4*)&Ctx[((size_t)b_ * SEQ + (q0 + r)) * DMODEL + h_ * 64 + c * 8] = val;
  }
}

// ---------------------------------------------------------------------------
// Kernel 3: output projection  out = Ctx @ Wo^T + bo   (fp32 result)
// A-tile is bf16 and copy-through -> async global->LDS when available.
// ---------------------------------------------------------------------------
__global__ __launch_bounds__(256) void outproj_kernel(
    const unsigned short* __restrict__ Ctx,  // [B*S, D] bf16
    const float* __restrict__ Wo,            // [D, D] fp32
    const float* __restrict__ bo,            // [D]
    float* __restrict__ Out)                 // [B*S, D] fp32
{
  __shared__ __align__(16) unsigned short As[128][72];
  __shared__ __align__(16) unsigned short Bs[64][72];

  const int tid  = threadIdx.x;
  const int wave = tid >> 5;
  const int lane = tid & 31;
  const int m0   = blockIdx.x * 128;
  const int n0   = blockIdx.y * 64;

  v8f acc[4] = {};

  for (int k0 = 0; k0 < DMODEL; k0 += 64) {
    __syncthreads();
#if USE_ASYNC
#pragma unroll
    for (int j = 0; j < 4; ++j) {            // 128 rows x 8 chunks = 1024 chunks
      const int item = j * 256 + tid;
      const int r = item >> 3, c = item & 7;
      async_copy16(&Ctx[(size_t)(m0 + r) * DMODEL + k0 + c * 8], &As[r][c * 8]);
    }
#else
    for (int idx = tid; idx < 128 * 64; idx += 256) {
      int r = idx >> 6, c = idx & 63;
      As[r][c] = Ctx[(size_t)(m0 + r) * DMODEL + k0 + c];
    }
#endif
    for (int idx = tid; idx < 64 * 64; idx += 256) {
      int r = idx >> 6, c = idx & 63;
      Bs[r][c] = f32_to_bf16(Wo[(size_t)(n0 + r) * DMODEL + k0 + c]);
    }
#if USE_ASYNC
    __builtin_amdgcn_s_wait_asynccnt(0);
#endif
    __syncthreads();

#pragma unroll
    for (int kk = 0; kk < 64; kk += 32) {
      v16bf a = load_frag_a<72>(&As[0][0], wave * 16, kk);
#pragma unroll
      for (int nt = 0; nt < 4; ++nt)
        acc[nt] = wmma_bf16(a, load_frag_b<72>(&Bs[0][0], nt * 16, kk), acc[nt]);
    }
  }

  const int col   = lane & 15;
  const int rbase = (lane >> 4) * 8;
#pragma unroll
  for (int nt = 0; nt < 4; ++nt) {
    const int n  = n0 + nt * 16 + col;
    const float bv = bo[n];
#pragma unroll
    for (int r = 0; r < 8; ++r) {
      const int m = m0 + wave * 16 + rbase + r;
      Out[(size_t)m * DMODEL + n] = acc[nt][r] + bv;
    }
  }
}

// ---------------------------------------------------------------------------
extern "C" void kernel_launch(void* const* d_in, const int* in_sizes, int n_in,
                              void* d_out, int out_size, void* d_ws, size_t ws_size,
                              hipStream_t stream) {
  (void)in_sizes; (void)n_in; (void)out_size; (void)ws_size;

  const float* q    = (const float*)d_in[0];
  const float* k    = (const float*)d_in[1];
  const float* v    = (const float*)d_in[2];
  const int*   mask = (const int*)d_in[3];
  const float* Wq = (const float*)d_in[4];  const float* bq = (const float*)d_in[5];
  const float* Wk = (const float*)d_in[6];  const float* bk = (const float*)d_in[7];
  const float* Wv = (const float*)d_in[8];  const float* bv = (const float*)d_in[9];
  const float* Wo = (const float*)d_in[10]; const float* bo = (const float*)d_in[11];

  const size_t HSZ = (size_t)BATCH * NHEAD * SEQ * DK;   // 4M bf16 elements
  unsigned short* Qp  = (unsigned short*)d_ws;           // [B,H,S,DK]
  unsigned short* Kp  = Qp + HSZ;                        // [B,H,S,DK]
  unsigned short* Vp  = Kp + HSZ;                        // [B,H,DK,S]
  unsigned short* Ctx = Vp + HSZ;                        // [B,S,D]

  const dim3 blk(256);
  const dim3 gProj((BATCH * SEQ) / 128, DMODEL / 64);    // 32 x 16
  const dim3 gAttn(SEQ / 128, BATCH * NHEAD);            // 16 x 32

  proj_bf16_kernel<false><<<gProj, blk, 0, stream>>>(q, Wq, bq, Qp);
  proj_bf16_kernel<false><<<gProj, blk, 0, stream>>>(k, Wk, bk, Kp);
  proj_bf16_kernel<true ><<<gProj, blk, 0, stream>>>(v, Wv, bv, Vp);
  flash_attn_kernel<<<gAttn, blk, 0, stream>>>(Qp, Kp, Vp, mask, Ctx);
  outproj_kernel<<<gProj, blk, 0, stream>>>(Ctx, Wo, bo, (float*)d_out);
}
